// HybridAttention_48550310314423
// MI455X (gfx1250) — matrix operand order repaired and checked by reading, not compiled, not run on hardware
//
#include <hip/hip_runtime.h>
#include <hip/hip_bf16.h>

#define HID 2048
#define BATCH 2
#define SEQ 2048
#define NH 16
#define HD 128
#define MROWS (BATCH * SEQ)          // 4096
#define SCALE_ 0.08838834764831845f  // 1/sqrt(128)
#define LOG2E_ 1.4426950408889634f

typedef __attribute__((ext_vector_type(16))) __bf16 v16bf;
typedef __attribute__((ext_vector_type(8)))  float  v8f;
typedef __attribute__((ext_vector_type(4)))  unsigned int u32x4;
typedef __attribute__((ext_vector_type(8)))  int i32x8;
typedef __attribute__((ext_vector_type(4)))  int i32x4;

#if defined(__HIP_DEVICE_COMPILE__) && __has_builtin(__builtin_amdgcn_tensor_load_to_lds)
#define HAVE_TDM 1
#else
#define HAVE_TDM 0
#endif

__device__ __forceinline__ v8f zero8() {
    v8f z = {0.f, 0.f, 0.f, 0.f, 0.f, 0.f, 0.f, 0.f};
    return z;
}

// A fragment: 16(M) x 32(K) bf16, src row-major (m,k) at m*ld+k.
// ISA 7.12.2: lanes 0-15 hold M=lane; elems 0..7 -> K=8*half+0..7,
//             elems 8..15 -> K=8*half+16..23.
__device__ __forceinline__ v16bf frag_a(const __bf16* src, int ld, int lane) {
    const int half = lane >> 4, m = lane & 15;
    const __bf16* p = src + m * ld + 8 * half;
    v16bf a;
    ((int4*)&a)[0] = *(const int4*)(p);
    ((int4*)&a)[1] = *(const int4*)(p + 16);
    return a;
}

// B fragment: 32(K) x 16(N) bf16, srcT stored N-major: element (k,n) at n*ld+k.
// ISA: lane = column n (mod 16); elems 0..7 -> K=8*half+0..7, elems 8..15 -> K=8*half+16..23.
__device__ __forceinline__ v16bf frag_b(const __bf16* srcT, int ld, int lane) {
    const int half = lane >> 4, n = lane & 15;
    const __bf16* p = srcT + n * ld + 8 * half;
    v16bf b;
    ((int4*)&b)[0] = *(const int4*)(p);
    ((int4*)&b)[1] = *(const int4*)(p + 16);
    return b;
}

__device__ __forceinline__ v8f wmma_bf16(v16bf a, v16bf b, v8f c) {
    return __builtin_amdgcn_wmma_f32_16x16x32_bf16(false, a, false, b,
                                                   (short)0, c, false, false);
}

// ---------------------------------------------------------------------------
// Tensor Data Mover: 2D bf16 tile (tile_d0 x tile_d1 elems, row stride
// stride_d0 elems) global -> LDS. Descriptor per ISA 8.3/8.4; type=2 (image),
// count=1, data_size code 1 (2 bytes). 6-arg builtin (clang-23 form).
// ---------------------------------------------------------------------------
#if HAVE_TDM
__device__ __forceinline__ void tdm_load_2d(unsigned lds_addr,
                                            unsigned long long gaddr,
                                            int tile_d0, int tile_d1,
                                            int stride_d0) {
    u32x4 g0;
    g0[0] = 1u;                                            // count=1, user mode
    g0[1] = lds_addr;                                      // lds_addr[63:32]
    g0[2] = (unsigned)(gaddr & 0xFFFFFFFFull);             // global_addr lo
    g0[3] = (unsigned)((gaddr >> 32) & 0x1FFFFFFull) | (2u << 30);  // hi | type=2
    i32x8 g1;
    g1[0] = (1 << 16);                                     // data_size = 2B
    g1[1] = (tile_d0 & 0xFFFF) << 16;                      // tensor_dim0[15:0]
    g1[2] = (tile_d1 & 0xFFFF) << 16;                      // tensor_dim1[15:0]
    g1[3] = (tile_d0 & 0xFFFF) << 16;                      // tile_dim0
    g1[4] = (tile_d1 & 0xFFFF);                            // tile_dim1
    g1[5] = stride_d0;                                     // dim0_stride[31:0]
    g1[6] = 0;
    g1[7] = 0;
    i32x4 gz4 = {0, 0, 0, 0};
    i32x8 gz8 = {0, 0, 0, 0, 0, 0, 0, 0};
    __builtin_amdgcn_tensor_load_to_lds(g0, g1, gz4, gz4, gz8, 0);
}
#endif

// ---------------------------------------------------------------------------
// Projection GEMM: Y[m][n] = sum_k X[m][k]*W[n][k]; X source per head
// (head = n>>7: <8 -> Xa, else Xb). Output bf16 head-major (b,h,s,d).
// Block tile 128x128 (one full head wide), 4 waves in 2x2; each wave owns a
// 64x64 quadrant -> 16 WMMAs per 16 ds_load_b128 per K-step (K-step = 32).
// LDS: 2x8KB staging overlaid with 32KB C-repack buffer.
// ---------------------------------------------------------------------------
__global__ __launch_bounds__(128) void proj_kernel(const float* __restrict__ Xa,
                                                   const float* __restrict__ Xb,
                                                   const float* __restrict__ W,
                                                   __bf16* __restrict__ Y) {
    __shared__ __align__(16) __bf16 SM[128 * 128];  // 32 KB
    __bf16(*Xs)[32] = reinterpret_cast<__bf16(*)[32]>(SM);          // 128x32
    __bf16(*Ws)[32] = reinterpret_cast<__bf16(*)[32]>(SM + 4096);   // 128x32
    __bf16(*Cs)[128] = reinterpret_cast<__bf16(*)[128]>(SM);        // 128x128

    const int tile_n = blockIdx.x * 128;    // == head * 128
    const int tile_m = blockIdx.y * 128;
    const int head = tile_n >> 7;
    const float* __restrict__ X = (head < (NH / 2)) ? Xa : Xb;
    const int t = threadIdx.x;
    const int lane = t & 31, w = t >> 5;
    const int wm = w >> 1, wn = w & 1;      // wave quadrant in 2x2
    const int srow = t >> 2, schunk = (t & 3) * 8;  // staging: 4 threads/row

    v8f acc[4][4];
#pragma unroll
    for (int i = 0; i < 4; i++)
#pragma unroll
        for (int j = 0; j < 4; j++) acc[i][j] = zero8();

    for (int k0 = 0; k0 < HID; k0 += 32) {
        __syncthreads();
        // stage X[128x32] and W[128x32] as bf16 (4 lanes cover 128B per row)
#pragma unroll
        for (int p = 0; p < 4; p++) {
            const int rr = srow + 32 * p;
            const float* gx = X + (size_t)(tile_m + rr) * HID + k0 + schunk;
            const float* gw = W + (size_t)(tile_n + rr) * HID + k0 + schunk;
            float4 x0 = ((const float4*)gx)[0];
            float4 x1 = ((const float4*)gx)[1];
            float4 w0 = ((const float4*)gw)[0];
            float4 w1 = ((const float4*)gw)[1];
            Xs[rr][schunk + 0] = (__bf16)x0.x;
            Xs[rr][schunk + 1] = (__bf16)x0.y;
            Xs[rr][schunk + 2] = (__bf16)x0.z;
            Xs[rr][schunk + 3] = (__bf16)x0.w;
            Xs[rr][schunk + 4] = (__bf16)x1.x;
            Xs[rr][schunk + 5] = (__bf16)x1.y;
            Xs[rr][schunk + 6] = (__bf16)x1.z;
            Xs[rr][schunk + 7] = (__bf16)x1.w;
            Ws[rr][schunk + 0] = (__bf16)w0.x;
            Ws[rr][schunk + 1] = (__bf16)w0.y;
            Ws[rr][schunk + 2] = (__bf16)w0.z;
            Ws[rr][schunk + 3] = (__bf16)w0.w;
            Ws[rr][schunk + 4] = (__bf16)w1.x;
            Ws[rr][schunk + 5] = (__bf16)w1.y;
            Ws[rr][schunk + 6] = (__bf16)w1.z;
            Ws[rr][schunk + 7] = (__bf16)w1.w;
        }
        __syncthreads();
        v16bf a[4], b[4];
#pragma unroll
        for (int i = 0; i < 4; i++)
            a[i] = frag_a(&Xs[64 * wm + 16 * i][0], 32, lane);
#pragma unroll
        for (int j = 0; j < 4; j++)
            b[j] = frag_b(&Ws[64 * wn + 16 * j][0], 32, lane);
#pragma unroll
        for (int i = 0; i < 4; i++)
#pragma unroll
            for (int j = 0; j < 4; j++)
                acc[i][j] = wmma_bf16(a[i], b[j], acc[i][j]);
    }

    // Repack C (128x128 bf16, overlaid on staging), then one contiguous
    // 256B row store per thread in head-major layout.
    __syncthreads();
    const int half = lane >> 4, col = lane & 15;
#pragma unroll
    for (int i = 0; i < 4; i++)
#pragma unroll
        for (int j = 0; j < 4; j++)
#pragma unroll
            for (int r = 0; r < 8; r++)
                Cs[64 * wm + 16 * i + r + 8 * half][64 * wn + 16 * j + col] =
                    (__bf16)acc[i][j][r];
    __syncthreads();
    {
        const int m = tile_m + t;
        const int bb = m / SEQ, s = m % SEQ;
        __bf16* dst = Y + (((size_t)bb * NH + head) * SEQ + s) * HD;
#pragma unroll
        for (int j = 0; j < 16; j++)
            ((int4*)dst)[j] = ((const int4*)&Cs[t][0])[j];
    }
}

// ---------------------------------------------------------------------------
// Flash attention: block = (b*h, 64 q-rows); 4 waves x 16 q-rows.
// K-tile = 64 keys; QK^T and P.V on bf16 WMMA, f32 online softmax.
// Q and K tiles staged by the Tensor Data Mover.
// ---------------------------------------------------------------------------
__global__ __launch_bounds__(128) void attn_kernel(const __bf16* __restrict__ Qw,
                                                   const __bf16* __restrict__ Kw,
                                                   const __bf16* __restrict__ Vw,
                                                   __bf16* __restrict__ Ow) {
    __shared__ __align__(16) __bf16 Qs[64][HD];     // 16 KB (O repack at end)
    __shared__ __align__(16) __bf16 Ks[64][HD];     // 16 KB
    __shared__ __align__(16) __bf16 Vt[HD][64];     // 16 KB (transposed)
    __shared__ __align__(16) __bf16 Ps[4][16][64];  // 8 KB (per-wave P tile)

    const int bh = blockIdx.y;
    const int q0 = blockIdx.x * 64;
    const size_t base = (size_t)bh * SEQ * HD;
    const int t = threadIdx.x, lane = t & 31, w = t >> 5;
    const int lr = t >> 1, lh = t & 1;
    const int half = lane >> 4, col = lane & 15;

    // ---- stage Q tile ----
#if HAVE_TDM
    if (w == 0) {
        tdm_load_2d((unsigned)(size_t)&Qs[0][0],
                    (unsigned long long)(size_t)(Qw + base + (size_t)q0 * HD),
                    HD, 64, HD);
        __builtin_amdgcn_s_wait_tensorcnt(0);
    }
#else
    {
        const __bf16* gq = Qw + base + (size_t)(q0 + lr) * HD + 64 * lh;
#pragma unroll
        for (int j = 0; j < 8; j++)
            ((int4*)&Qs[lr][64 * lh])[j] = ((const int4*)gq)[j];
    }
#endif
    __syncthreads();

    // Q fragments are invariant over the whole key loop: load once.
    v16bf aq[4];
#pragma unroll
    for (int ks = 0; ks < 4; ks++) aq[ks] = frag_a(&Qs[16 * w][32 * ks], HD, lane);

    float mrow[8], lrow[8];
    v8f Oacc[8];
#pragma unroll
    for (int r = 0; r < 8; r++) { mrow[r] = -1e30f; lrow[r] = 0.f; }
#pragma unroll
    for (int dt = 0; dt < 8; dt++) Oacc[dt] = zero8();

    for (int kt = 0; kt < SEQ; kt += 64) {
        __syncthreads();
        // K tile: async TDM (wave 0) or manual copy
#if HAVE_TDM
        if (w == 0)
            tdm_load_2d((unsigned)(size_t)&Ks[0][0],
                        (unsigned long long)(size_t)(Kw + base + (size_t)kt * HD),
                        HD, 64, HD);
#else
        {
            const __bf16* gk = Kw + base + (size_t)(kt + lr) * HD + 64 * lh;
#pragma unroll
            for (int j = 0; j < 8; j++)
                ((int4*)&Ks[lr][64 * lh])[j] = ((const int4*)gk)[j];
        }
#endif
        // V tile, transposed -> Vt[d][key] (feeds contiguous B fragments)
        {
            const __bf16* gv = Vw + base + (size_t)(kt + lr) * HD + 64 * lh;
            if (kt + 64 < SEQ) __builtin_prefetch(gv + 64 * HD, 0, 0);
#pragma unroll
            for (int j = 0; j < 8; j++) {
                int4 chunk = ((const int4*)gv)[j];
                const __bf16* cv = (const __bf16*)&chunk;
#pragma unroll
                for (int e = 0; e < 8; e++)
                    Vt[64 * lh + 8 * j + e][lr] = cv[e];
            }
        }
#if HAVE_TDM
        if (w == 0) __builtin_amdgcn_s_wait_tensorcnt(0);
#endif
        __syncthreads();

        // scores: 4 key-subtiles of 16; contraction over d=128 (4 WMMAs each)
        v8f sc[4];
#pragma unroll
        for (int st = 0; st < 4; st++) {
            v16bf bb[4];
#pragma unroll
            for (int ks = 0; ks < 4; ks++)
                bb[ks] = frag_b(&Ks[16 * st][32 * ks], HD, lane);
            v8f s = zero8();
#pragma unroll
            for (int ks = 0; ks < 4; ks++) s = wmma_bf16(aq[ks], bb[ks], s);
            sc[st] = s;
        }

        // online softmax (row = r + 8*half within this wave's 16-row slab)
#pragma unroll
        for (int r = 0; r < 8; r++) {
            float mx = fmaxf(fmaxf(sc[0][r], sc[1][r]),
                             fmaxf(sc[2][r], sc[3][r])) * SCALE_;
#pragma unroll
            for (int off = 1; off < 16; off <<= 1)
                mx = fmaxf(mx, __shfl_xor(mx, off, 32));
            const float mnew = fmaxf(mrow[r], mx);
            const float corr = exp2f((mrow[r] - mnew) * LOG2E_);
            mrow[r] = mnew;
            float rs = 0.f;
#pragma unroll
            for (int st = 0; st < 4; st++) {
                float p = exp2f((sc[st][r] * SCALE_ - mnew) * LOG2E_);
                sc[st][r] = p;
                rs += p;
            }
#pragma unroll
            for (int off = 1; off < 16; off <<= 1)
                rs += __shfl_xor(rs, off, 32);
            lrow[r] = lrow[r] * corr + rs;
#pragma unroll
            for (int dt = 0; dt < 8; dt++) Oacc[dt][r] *= corr;
        }

        // P (C layout) -> per-wave LDS tile; wave-local LDS is in-order
#pragma unroll
        for (int st = 0; st < 4; st++)
#pragma unroll
            for (int r = 0; r < 8; r++)
                Ps[w][r + 8 * half][st * 16 + col] = (__bf16)sc[st][r];

        // O += P @ V: contraction over 64 keys (2 chunks), 8 d-subtiles
        v16bf pa[2];
#pragma unroll
        for (int kc = 0; kc < 2; kc++)
            pa[kc] = frag_a(&Ps[w][0][32 * kc], 64, lane);
#pragma unroll
        for (int dt = 0; dt < 8; dt++) {
            v16bf b0 = frag_b(&Vt[16 * dt][0], 64, lane);
            v16bf b1 = frag_b(&Vt[16 * dt][32], 64, lane);
            Oacc[dt] = wmma_bf16(pa[0], b0, Oacc[dt]);
            Oacc[dt] = wmma_bf16(pa[1], b1, Oacc[dt]);
        }
    }

    // normalize, repack through Qs (wave-local rows), coalesced b128 stores
    float rinv[8];
#pragma unroll
    for (int r = 0; r < 8; r++) rinv[r] = 1.0f / lrow[r];
#pragma unroll
    for (int dt = 0; dt < 8; dt++)
#pragma unroll
        for (int r = 0; r < 8; r++)
            Qs[16 * w + r + 8 * half][dt * 16 + col] = (__bf16)(Oacc[dt][r] * rinv[r]);
    {
        const int orow = 16 * w + (lane >> 1);
        const int oh = lane & 1;
        __bf16* dst = Ow + base + (size_t)(q0 + orow) * HD + 64 * oh;
#pragma unroll
        for (int j = 0; j < 8; j++)
            ((int4*)dst)[j] = ((const int4*)&Qs[orow][64 * oh])[j];
    }
}

// ---------------------------------------------------------------------------
// Output projection: out[m][n] = sum_k A[m][k]*Wo[n][k]; A bf16 head-major
// (k = h*128+d), out f32 row-major. C repacked through LDS for coalescing.
// ---------------------------------------------------------------------------
__global__ __launch_bounds__(128) void outproj_kernel(const __bf16* __restrict__ Aw,
                                                      const float* __restrict__ W,
                                                      float* __restrict__ out) {
    __shared__ __align__(16) __bf16 As[64][32];
    __shared__ __align__(16) __bf16 Ws[64][32];
    __shared__ __align__(16) float Cf[64][64];
    const int tile_n = blockIdx.x * 64;
    const int tile_m = blockIdx.y * 64;
    const int t = threadIdx.x;
    const int lane = t & 31, w = t >> 5;
    const int lr = t >> 1, lh = t & 1;

    v8f acc[4];
#pragma unroll
    for (int i = 0; i < 4; i++) acc[i] = zero8();

    for (int k0 = 0; k0 < HID; k0 += 32) {
        __syncthreads();
        {
            const int k = k0 + 16 * lh;  // 16-wide chunk stays within one head
            const int m = tile_m + lr;
            const int bb = m / SEQ, s = m % SEQ;
            const int h = k >> 7, dd = k & (HD - 1);
            const __bf16* ga = Aw + (((size_t)bb * NH + h) * SEQ + s) * HD + dd;
            ((int4*)&As[lr][16 * lh])[0] = ((const int4*)ga)[0];
            ((int4*)&As[lr][16 * lh])[1] = ((const int4*)ga)[1];

            const float* gw = W + (size_t)(tile_n + lr) * HID + k0 + 16 * lh;
#pragma unroll
            for (int j = 0; j < 16; j += 4) {
                float4 vw = *(const float4*)(gw + j);
                Ws[lr][16 * lh + j + 0] = (__bf16)vw.x;
                Ws[lr][16 * lh + j + 1] = (__bf16)vw.y;
                Ws[lr][16 * lh + j + 2] = (__bf16)vw.z;
                Ws[lr][16 * lh + j + 3] = (__bf16)vw.w;
            }
        }
        __syncthreads();
        v16bf a = frag_a(&As[16 * w][0], 32, lane);
        v16bf bb[4];
#pragma unroll
        for (int nt = 0; nt < 4; nt++) bb[nt] = frag_b(&Ws[16 * nt][0], 32, lane);
#pragma unroll
        for (int nt = 0; nt < 4; nt++) acc[nt] = wmma_bf16(a, bb[nt], acc[nt]);
    }

    const int half = lane >> 4, col = lane & 15;
#pragma unroll
    for (int nt = 0; nt < 4; nt++)
#pragma unroll
        for (int r = 0; r < 8; r++)
            Cf[16 * w + r + 8 * half][nt * 16 + col] = acc[nt][r];
    __syncthreads();
    {
        const int m = tile_m + lr;
        float* dst = out + (size_t)m * HID + tile_n + 32 * lh;
#pragma unroll
        for (int j = 0; j < 8; j++)
            ((float4*)dst)[j] = ((const float4*)&Cf[lr][32 * lh])[j];
    }
}

extern "C" void kernel_launch(void* const* d_in, const int* in_sizes, int n_in,
                              void* d_out, int out_size, void* d_ws, size_t ws_size,
                              hipStream_t stream) {
    const float* x_q  = (const float*)d_in[0];
    const float* x_kv = (const float*)d_in[1];
    const float* Wq   = (const float*)d_in[2];
    const float* Wk   = (const float*)d_in[3];
    const float* Wv   = (const float*)d_in[4];
    const float* Wo   = (const float*)d_in[5];
    float* out = (float*)d_out;

    const size_t elems = (size_t)MROWS * HID;  // 8M bf16 per buffer
    __bf16* qws = (__bf16*)d_ws;
    __bf16* kws = qws + elems;
    __bf16* vws = kws + elems;
    __bf16* aws = vws + elems;

    dim3 blk(128);

    // Q from x_q (all heads); K,V: heads 0-7 from x_q (self), 8-15 from x_kv
    dim3 gproj(HID / 128, MROWS / 128);
    proj_kernel<<<gproj, blk, 0, stream>>>(x_q, x_q, Wq, qws);
    proj_kernel<<<gproj, blk, 0, stream>>>(x_q, x_kv, Wk, kws);
    proj_kernel<<<gproj, blk, 0, stream>>>(x_q, x_kv, Wv, vws);

    attn_kernel<<<dim3(SEQ / 64, BATCH * NH), blk, 0, stream>>>(qws, kws, vws, aws);

    outproj_kernel<<<dim3(HID / 64, MROWS / 64), blk, 0, stream>>>(aws, Wo, out);
}
